// RetICLBase_24567212933152
// MI455X (gfx1250) — matrix operand order repaired and checked by reading, not compiled, not run on hardware
//
#include <hip/hip_runtime.h>

// ---------------------------------------------------------------------------
// RetICL forward for MI455X (gfx1250, wave32)
//   B=256, T=8, E=H=768, N=20000
//   - all GEMMs via v_wmma_f32_16x16x32_bf16, fp32 accumulate
//   - global->LDS staging via global_load_async_to_lds_b128 (ASYNCcnt),
//     double-buffered LDS, branch-free steady-state K loop + peeled last tile
//   - scatter-mask scan fused into the scoring GEMM epilogue
//   - non-temporal stores for the 164MB activations (keeps corpus in L2)
// ---------------------------------------------------------------------------

typedef __attribute__((ext_vector_type(16))) __bf16 v16bf;
typedef __attribute__((ext_vector_type(8)))  float  v8f;

#define Bb   256
#define Tt   8
#define Ee   768
#define Hh   768
#define Nn   20000
#define S3H  2304          // 3*H
#define BH   (Bb*Hh)

__device__ __forceinline__ __bf16 f2bf(float f) { return (__bf16)f; }

// ---------------------------------------------------------------------------
// conversion kernels (one-time, tiny vs main GEMM)
// ---------------------------------------------------------------------------
__global__ void cvt_copy_bf16(const float* __restrict__ s, __bf16* __restrict__ d, int n) {
    for (int i = blockIdx.x * blockDim.x + threadIdx.x; i < n; i += gridDim.x * blockDim.x)
        d[i] = f2bf(s[i]);
}

// dst[c][r] = src[r][c];  src is [R][C] f32, dst is [C][R] bf16
__global__ void cvt_transpose_bf16(const float* __restrict__ s, __bf16* __restrict__ d, int R, int C) {
    int total = R * C;
    for (int i = blockIdx.x * blockDim.x + threadIdx.x; i < total; i += gridDim.x * blockDim.x) {
        int c = i / R, r = i % R;
        d[i] = f2bf(s[(size_t)r * C + c]);
    }
}

// xs[t*B + b][k] = examples[b][t][k], t in [0,T-1)
__global__ void cvt_examples_bf16(const float* __restrict__ ex, __bf16* __restrict__ d) {
    const int total = (Tt - 1) * Bb * Ee;
    for (int i = blockIdx.x * blockDim.x + threadIdx.x; i < total; i += gridDim.x * blockDim.x) {
        int row = i / Ee, k = i - row * Ee;
        int t = row >> 8, b = row & 255;
        d[i] = f2bf(ex[((size_t)(b * Tt + t)) * Ee + k]);
    }
}

// ---------------------------------------------------------------------------
// bf16 GEMM:  C[M][N] = A[M][K=768] * Bm[N][K=768]^T  (fp32 accum via WMMA)
// block tile 64x128, 256 threads = 8 waves, wave tile 16x64 (4 accumulators)
// MODE 0: store f32            MODE 1: store bf16
// MODE 2: tanh(acc+bias[n]) -> f32 + bf16 (GRU h0)
// MODE 3: scoring: remap row t*B+b -> b*T+t, -inf where previously used, NT store
// M must be a multiple of 64; N edge handled (N=20000)
// ---------------------------------------------------------------------------
#define TILE_K     32
#define LDS_STRIDE 40            // 32 + 8 pad bf16 -> conflict-free 16B lanes
#define NKT        24            // 768 / 32
#define ABUF_B     (64  * LDS_STRIDE * 2)   // bytes per A buffer
#define BBUF_B     (128 * LDS_STRIDE * 2)   // bytes per B buffer

template <int MODE>
__global__ __launch_bounds__(256) void gemm_bf16_kernel(
    const __bf16* __restrict__ A, const __bf16* __restrict__ Bm,
    int N,
    float* __restrict__ Cf, __bf16* __restrict__ Cbf,
    const float* __restrict__ bias, const int* __restrict__ idx)
{
    constexpr int K = 768;
    __shared__ __bf16 As[2 * 64 * LDS_STRIDE];
    __shared__ __bf16 Bs[2 * 128 * LDS_STRIDE];

    const int tid  = threadIdx.x;
    const int lane = tid & 31;
    const int wave = tid >> 5;
    const int wm   = wave & 3;   // 4 m-subtiles of 16
    const int wn   = wave >> 2;  // 2 n-subtiles of 64

    const int m0 = blockIdx.y * 64;
    const int n0 = blockIdx.x * 128;

    // ---- cooperative async loaders: A 64x32 (1 op/thread), B 128x32 (2 ops) --
    const int lrow = tid >> 2;           // 0..63
    const int lchk = (tid & 3) * 8;      // bf16 elems within the 32-wide K slab
    const int arow = m0 + lrow;
    int br0 = n0 + lrow;       if (br0 >= N) br0 = N - 1;
    int br1 = n0 + 64 + lrow;  if (br1 >= N) br1 = N - 1;

    unsigned a_go  = ((unsigned)arow * K + (unsigned)lchk) * 2u;   // byte offsets
    unsigned b_go0 = ((unsigned)br0  * K + (unsigned)lchk) * 2u;
    unsigned b_go1 = ((unsigned)br1  * K + (unsigned)lchk) * 2u;
    const unsigned long long Abase = (unsigned long long)(size_t)A;
    const unsigned long long Bbase = (unsigned long long)(size_t)Bm;

    const unsigned aW  = (unsigned)(size_t)As + (unsigned)(lrow * (LDS_STRIDE * 2) + lchk * 2);
    const unsigned bW0 = (unsigned)(size_t)Bs + (unsigned)(lrow * (LDS_STRIDE * 2) + lchk * 2);
    const unsigned bW1 = (unsigned)(size_t)Bs + (unsigned)((64 + lrow) * (LDS_STRIDE * 2) + lchk * 2);

    auto issue_tile = [&](int buf) {
        unsigned ao  = aW  + (unsigned)buf * ABUF_B;
        unsigned bo0 = bW0 + (unsigned)buf * BBUF_B;
        unsigned bo1 = bW1 + (unsigned)buf * BBUF_B;
        asm volatile("global_load_async_to_lds_b128 %0, %1, %2"
                     :: "v"(ao),  "v"(a_go),  "s"(Abase) : "memory");
        asm volatile("global_load_async_to_lds_b128 %0, %1, %2"
                     :: "v"(bo0), "v"(b_go0), "s"(Bbase) : "memory");
        asm volatile("global_load_async_to_lds_b128 %0, %1, %2"
                     :: "v"(bo1), "v"(b_go1), "s"(Bbase) : "memory");
        a_go += 64; b_go0 += 64; b_go1 += 64;   // next 32-wide K slab
    };

    v8f acc0 = {}, acc1 = {}, acc2 = {}, acc3 = {};

    const int fr    = lane & 15;
    const int khalf = (lane >> 4) * 16;  // A frag: K 0-15 vs 16-31 contiguous
    const int kb    = (lane >> 4) * 16;  // B frag: K 0-15 vs 16-31 contiguous

    const unsigned aR  = (unsigned)((wm * 16 + fr) * LDS_STRIDE);
    const unsigned bR  = (unsigned)((wn * 64 + fr) * LDS_STRIDE);

    auto compute_tile = [&](int buf) {
        const __bf16* Atile = As + buf * (64  * LDS_STRIDE) + aR;
        const __bf16* Btile = Bs + buf * (128 * LDS_STRIDE) + bR;
        union { v16bf v; uint4 u[2]; } fa, fb0, fb1, fb2, fb3;
        fa.u[0]  = *(const uint4*)(Atile + khalf);
        fa.u[1]  = *(const uint4*)(Atile + khalf + 8);
        fb0.u[0] = *(const uint4*)(Btile + kb);
        fb0.u[1] = *(const uint4*)(Btile + kb + 8);
        fb1.u[0] = *(const uint4*)(Btile + 16 * LDS_STRIDE + kb);
        fb1.u[1] = *(const uint4*)(Btile + 16 * LDS_STRIDE + kb + 8);
        fb2.u[0] = *(const uint4*)(Btile + 32 * LDS_STRIDE + kb);
        fb2.u[1] = *(const uint4*)(Btile + 32 * LDS_STRIDE + kb + 8);
        fb3.u[0] = *(const uint4*)(Btile + 48 * LDS_STRIDE + kb);
        fb3.u[1] = *(const uint4*)(Btile + 48 * LDS_STRIDE + kb + 8);
        acc0 = __builtin_amdgcn_wmma_f32_16x16x32_bf16(false, fa.v, false, fb0.v,
                                                       (short)0, acc0, false, false);
        acc1 = __builtin_amdgcn_wmma_f32_16x16x32_bf16(false, fa.v, false, fb1.v,
                                                       (short)0, acc1, false, false);
        acc2 = __builtin_amdgcn_wmma_f32_16x16x32_bf16(false, fa.v, false, fb2.v,
                                                       (short)0, acc2, false, false);
        acc3 = __builtin_amdgcn_wmma_f32_16x16x32_bf16(false, fa.v, false, fb3.v,
                                                       (short)0, acc3, false, false);
    };

    // prologue: tile 0 -> buffer 0
    issue_tile(0);

    // branch-free steady state: always prefetch the next tile
#pragma unroll 2
    for (int kt = 0; kt < NKT - 1; ++kt) {
        issue_tile((kt + 1) & 1);
        asm volatile("s_wait_asynccnt 0x3" ::: "memory");   // tile kt landed
        __syncthreads();
        compute_tile(kt & 1);
        __syncthreads();    // frag reads drained -> buffer reusable
    }

    // peeled last tile
    asm volatile("s_wait_asynccnt 0x0" ::: "memory");
    __syncthreads();
    compute_tile((NKT - 1) & 1);

    // ---- epilogue — C layout: VGPR r holds M = r + 8*(lane>>4), N = lane&15 --
    const int mrowBase = m0 + wm * 16 + 8 * (lane >> 4);
    const int nbase    = n0 + wn * 64 + (lane & 15);
    const v8f accs[4] = {acc0, acc1, acc2, acc3};

#pragma unroll
    for (int r = 0; r < 8; ++r) {
        const int m = mrowBase + r;
        if (MODE == 3) {   // scoring + fused scatter-mask, NT stores
            const int t = m >> 8;                  // gemm row = t*B + b
            const int b = m & 255;
            const size_t orow = (size_t)(b * Tt + t);
            int used[Tt - 1];
#pragma unroll
            for (int tp = 0; tp < Tt - 1; ++tp)
                used[tp] = (tp < t) ? idx[b * Tt + tp] : -1;
            const float ninf = -__builtin_inff();
#pragma unroll
            for (int j = 0; j < 4; ++j) {
                const int nc = nbase + j * 16;
                bool dead = false;
#pragma unroll
                for (int tp = 0; tp < Tt - 1; ++tp) dead |= (used[tp] == nc);
                if (nc < N)
                    __builtin_nontemporal_store(dead ? ninf : accs[j][r], &Cf[orow * N + nc]);
            }
        } else {
#pragma unroll
            for (int j = 0; j < 4; ++j) {
                const int nc = nbase + j * 16;
                if (nc >= N) continue;
                const float v = accs[j][r];
                if (MODE == 0) {
                    Cf[(size_t)m * N + nc] = v;
                } else if (MODE == 1) {
                    Cbf[(size_t)m * N + nc] = f2bf(v);
                } else { // MODE 2
                    const float y = tanhf(v + bias[nc]);
                    Cf[(size_t)m * N + nc]  = y;
                    Cbf[(size_t)m * N + nc] = f2bf(y);
                }
            }
        }
    }
}

// ---------------------------------------------------------------------------
// GRU gate fusion: h_t = (1-z)*n + z*h_{t-1}
// ---------------------------------------------------------------------------
__global__ __launch_bounds__(256) void gru_gate_kernel(
    const float* __restrict__ GXt, const float* __restrict__ GH,
    const float* __restrict__ bx,  const float* __restrict__ bh,
    const float* __restrict__ hprev, float* __restrict__ hout,
    __bf16* __restrict__ houtbf)
{
    int i = blockIdx.x * blockDim.x + threadIdx.x;
    if (i >= BH) return;
    int b = i / Hh, j = i - b * Hh;
    const size_t o = (size_t)b * S3H;
    float xr = GXt[o + j]          + bx[j];
    float xz = GXt[o + Hh + j]     + bx[Hh + j];
    float xn = GXt[o + 2 * Hh + j] + bx[2 * Hh + j];
    float hr = GH[o + j]           + bh[j];
    float hz = GH[o + Hh + j]      + bh[Hh + j];
    float hn = GH[o + 2 * Hh + j]  + bh[2 * Hh + j];
    float rg = 1.f / (1.f + __expf(-(xr + hr)));
    float zg = 1.f / (1.f + __expf(-(xz + hz)));
    float ng = tanhf(xn + rg * hn);
    float h  = (1.f - zg) * ng + zg * hprev[i];
    hout[i]   = h;
    houtbf[i] = f2bf(h);
}

// ---------------------------------------------------------------------------
// value estimates: out[b*T+t] = dot(latent[t][b], vfe_w) + vfe_b
// ---------------------------------------------------------------------------
__global__ __launch_bounds__(256) void value_kernel(
    const float* __restrict__ latent, const float* __restrict__ w,
    const float* __restrict__ vb, float* __restrict__ out)
{
    const int wave = threadIdx.x >> 5;
    const int lane = threadIdx.x & 31;
    const int row  = blockIdx.x * 8 + wave;    // row = t*B + b
    if (row >= Tt * Bb) return;
    const int t = row >> 8, b = row & 255;
    const float* lp = latent + (size_t)row * Hh;
    float s = 0.f;
    for (int k = lane; k < Hh; k += 32) s += lp[k] * w[k];
#pragma unroll
    for (int off = 16; off; off >>= 1) s += __shfl_down(s, off, 32);
    if (lane == 0) out[b * Tt + t] = s + vb[0];
}

// ---------------------------------------------------------------------------
// host launcher
// ---------------------------------------------------------------------------
static inline int ceil_div(int a, int b) { return (a + b - 1) / b; }

extern "C" void kernel_launch(void* const* d_in, const int* in_sizes, int n_in,
                              void* d_out, int out_size, void* d_ws, size_t ws_size,
                              hipStream_t stream)
{
    (void)in_sizes; (void)n_in; (void)out_size; (void)ws_size;
    const float* cur      = (const float*)d_in[0];   // [B,E]
    const float* examples = (const float*)d_in[1];   // [B,T,E]
    const float* corpus   = (const float*)d_in[2];   // [N,E]
    const int*   pidx     = (const int*)  d_in[3];   // [B,T]
    const float* W_init   = (const float*)d_in[4];   // [E,H]
    const float* b_init   = (const float*)d_in[5];   // [H]
    const float* Wx       = (const float*)d_in[6];   // [E,3H]
    const float* Wh       = (const float*)d_in[7];   // [H,3H]
    const float* bx       = (const float*)d_in[8];   // [3H]
    const float* bh       = (const float*)d_in[9];   // [3H]
    const float* bilinear = (const float*)d_in[10];  // [H,E]
    const float* vfe_w    = (const float*)d_in[11];  // [H]
    const float* vfe_b    = (const float*)d_in[12];  // [1]

    float* acts_out = (float*)d_out;                           // [B*T, N]
    float* ve_out   = acts_out + (size_t)Bb * Tt * Nn;         // [B*T]

    // workspace carve-out (~75 MB)
    char* base = (char*)d_ws;
    size_t off = 0;
    auto carve = [&](size_t bytes) { char* p = base + off; off += (bytes + 255) & ~(size_t)255; return p; };
    __bf16* corpusbf = (__bf16*)carve((size_t)Nn * Ee * 2);
    __bf16* curbf    = (__bf16*)carve((size_t)Bb * Ee * 2);
    __bf16* xsbf     = (__bf16*)carve((size_t)(Tt - 1) * Bb * Ee * 2);
    __bf16* WIT      = (__bf16*)carve((size_t)Hh * Ee * 2);
    __bf16* WxT      = (__bf16*)carve((size_t)S3H * Ee * 2);
    __bf16* WhT      = (__bf16*)carve((size_t)S3H * Hh * 2);
    __bf16* BT       = (__bf16*)carve((size_t)Ee * Hh * 2);
    float*  latent   = (float*) carve((size_t)Tt * BH * 4);    // [T][B][H]
    __bf16* latentbf = (__bf16*)carve((size_t)Tt * BH * 2);
    float*  GX       = (float*) carve((size_t)(Tt - 1) * Bb * S3H * 4);
    float*  GH       = (float*) carve((size_t)Bb * S3H * 4);
    __bf16* querybf  = (__bf16*)carve((size_t)Tt * Bb * Ee * 2);

    const dim3 blk(256);

    // --- one-time conversions ---
    cvt_copy_bf16<<<ceil_div(Nn * Ee, 256), blk, 0, stream>>>(corpus, corpusbf, Nn * Ee);
    cvt_copy_bf16<<<ceil_div(Bb * Ee, 256), blk, 0, stream>>>(cur, curbf, Bb * Ee);
    cvt_examples_bf16<<<ceil_div((Tt - 1) * Bb * Ee, 256), blk, 0, stream>>>(examples, xsbf);
    cvt_transpose_bf16<<<ceil_div(Ee * Hh, 256), blk, 0, stream>>>(W_init, WIT, Ee, Hh);
    cvt_transpose_bf16<<<ceil_div(Ee * S3H, 256), blk, 0, stream>>>(Wx, WxT, Ee, S3H);
    cvt_transpose_bf16<<<ceil_div(Hh * S3H, 256), blk, 0, stream>>>(Wh, WhT, Hh, S3H);
    cvt_transpose_bf16<<<ceil_div(Hh * Ee, 256), blk, 0, stream>>>(bilinear, BT, Hh, Ee);

    // --- h0 = tanh(cur @ W_init + b_init) ---
    gemm_bf16_kernel<2><<<dim3(Hh / 128, Bb / 64), blk, 0, stream>>>(
        curbf, WIT, Hh, latent, latentbf, b_init, nullptr);

    // --- GX = xs @ Wx (all steps at once) ---
    gemm_bf16_kernel<0><<<dim3(S3H / 128, (Tt - 1) * Bb / 64), blk, 0, stream>>>(
        xsbf, WxT, S3H, GX, nullptr, nullptr, nullptr);

    // --- GRU scan: GH = h_{t-1} @ Wh, then fused gates ---
    for (int t = 1; t < Tt; ++t) {
        gemm_bf16_kernel<0><<<dim3(S3H / 128, Bb / 64), blk, 0, stream>>>(
            latentbf + (size_t)(t - 1) * BH, WhT, S3H, GH, nullptr, nullptr, nullptr);
        gru_gate_kernel<<<ceil_div(BH, 256), blk, 0, stream>>>(
            GX + (size_t)(t - 1) * Bb * S3H, GH, bx, bh,
            latent + (size_t)(t - 1) * BH,
            latent + (size_t)t * BH, latentbf + (size_t)t * BH);
    }

    // --- query = latent @ bilinear (kept bf16 for scoring GEMM) ---
    gemm_bf16_kernel<1><<<dim3(Ee / 128, Tt * Bb / 64), blk, 0, stream>>>(
        latentbf, BT, Ee, nullptr, querybf, nullptr, nullptr);

    // --- activations = query @ corpus^T with fused scatter-mask epilogue ---
    gemm_bf16_kernel<3><<<dim3(ceil_div(Nn, 128), Tt * Bb / 64), blk, 0, stream>>>(
        querybf, corpusbf, Nn, acts_out, nullptr, nullptr, pidx);

    // --- value estimates ---
    value_kernel<<<ceil_div(Tt * Bb, 8), blk, 0, stream>>>(latent, vfe_w, vfe_b, ve_out);
}
